// CHUNKER_15375982920267
// MI455X (gfx1250) — compile-verified, matrix operand order checked by reading, not run on hardware
//
#include <hip/hip_runtime.h>

typedef __attribute__((ext_vector_type(16))) _Float16 v16h;
typedef __attribute__((ext_vector_type(8)))  float    v8f;
typedef __attribute__((ext_vector_type(4)))  unsigned int u32x4;

#define WDIM2 512          // 2*WDIM token feature width
#define HDIM  512
#define KREP  640          // HDIM + 4*POS_DIM
#define TOK_CHUNK 128
#define WAVES_PER_BLOCK 4
#define ROWS_PER_BLOCK 64

union FragH { v16h v; u32x4 q[2]; };
union Pack8 { _Float16 h[8]; u32x4 q; };

// LDS tile is [16 rows][640 halves], row stride 640 (1280 B). XOR-swizzle the
// 8-half chunk index by (row&7) so the 16-row A-fragment ds_load_b128s spread
// across banks instead of 16-way conflicting (640 halves = 320 dwords = 0 mod 64).
__device__ __forceinline__ int qoff(int row, int chunk) {
  return row * KREP + ((chunk ^ (row & 7)) << 3);
}
__device__ __forceinline__ int soff(int row, int col) {
  int chunk = col >> 3;
  return row * KREP + (((chunk ^ (row & 7)) << 3) | (col & 7));
}

// ---------------- prep 1: chunked inclusive scan of token embeddings -------
__global__ void scan_partial(const int* __restrict__ sentence, const int* __restrict__ pos_tags,
                             const float* __restrict__ We_wrd, const float* __restrict__ We_pos,
                             float* __restrict__ csum, float* __restrict__ ctot) {
  int d = threadIdx.x;           // 0..511 feature dim
  int c = blockIdx.x;            // 0..15 token chunk
  float run = 0.f;
  for (int i = 0; i < TOK_CHUNK; ++i) {
    int t = c * TOK_CHUNK + i;
    float v = (d < 256) ? We_pos[(size_t)pos_tags[t] * 256 + d]
                        : We_wrd[(size_t)sentence[t] * 256 + (d - 256)];
    run += v;
    csum[(size_t)(t + 1) * WDIM2 + d] = run;   // chunk-local inclusive for now
  }
  ctot[c * WDIM2 + d] = run;
}

__global__ void scan_fix(float* __restrict__ csum, const float* __restrict__ ctot) {
  int d = threadIdx.x;
  int c = blockIdx.x;
  if (c == 0) { csum[d] = 0.f; return; }
  float carry = 0.f;
  for (int p = 0; p < c; ++p) carry += ctot[p * WDIM2 + d];
  for (int i = 0; i < TOK_CHUNK; ++i)
    csum[(size_t)(c * TOK_CHUNK + i + 1) * WDIM2 + d] += carry;
}

// ---------------- prep 2: weights -> transposed f16 [n][k] -----------------
__global__ void wcvt(const float* __restrict__ w0, const float* __restrict__ w1,
                     const float* __restrict__ w2,
                     _Float16* __restrict__ Wt0, _Float16* __restrict__ Wt1,
                     _Float16* __restrict__ Wt2) {
  int i = blockIdx.x * blockDim.x + threadIdx.x;
  const int S = 512 * 512;
  if (i < S) {
    int n = i >> 9, k = i & 511;
    Wt0[(size_t)n * 512 + k] = (_Float16)w0[(size_t)k * 512 + n];
  } else if (i < 2 * S) {
    int j = i - S; int n = j >> 9, k = j & 511;
    Wt1[(size_t)n * 512 + k] = (_Float16)w1[(size_t)k * 512 + n];
  } else if (i < 2 * S + 512 * KREP) {
    int j = i - 2 * S; int n = j / KREP, k = j % KREP;
    Wt2[(size_t)n * KREP + k] = (_Float16)w2[(size_t)k * 512 + n];
  }
}

// ---------------- fused per-wave GEMM tile (relu, bias, f16 out to LDS) ----
__device__ __forceinline__ void gemm_tile(const _Float16* __restrict__ src,
                                          const _Float16* __restrict__ Wt, int K,
                                          const float* __restrict__ bias,
                                          _Float16* __restrict__ dst,
                                          int lrow, int lhi) {
  for (int nt = 0; nt < HDIM / 16; ++nt) {
    v8f c = {};
    int n = nt * 16 + lrow;
    const _Float16* wrow = Wt + (size_t)n * K;
    for (int ks = 0; ks < K; ks += 32) {
      FragH a, b;
      int c0 = (ks >> 3) + lhi;      // lane's K-half: lhi selects K offset +8
      a.q[0] = *(const u32x4*)(src + qoff(lrow, c0));
      a.q[1] = *(const u32x4*)(src + qoff(lrow, c0 + 2));
      b.q[0] = *(const u32x4*)(wrow + ks + lhi * 8);
      b.q[1] = *(const u32x4*)(wrow + ks + lhi * 8 + 16);
      c = __builtin_amdgcn_wmma_f32_16x16x32_f16(false, a.v, false, b.v,
                                                 (short)0, c, false, false);
    }
    float bb = bias[n];
    #pragma unroll
    for (int r = 0; r < 8; ++r) {
      int row = lhi * 8 + r;               // C layout: lanes>=16 start at M=8
      float v = c[r] + bb;
      dst[soff(row, n)] = (_Float16)(v > 0.f ? v : 0.f);
    }
  }
}

// ---------------- fused span pipeline --------------------------------------
__global__ __launch_bounds__(128) void fused_spans(
    const int* __restrict__ spans_start, const int* __restrict__ spans_len, int n_chunks,
    const float* __restrict__ csum,
    const _Float16* __restrict__ Wt0, const float* __restrict__ b0,
    const _Float16* __restrict__ Wt1, const float* __restrict__ b1,
    const _Float16* __restrict__ Wt2, const float* __restrict__ wsb0,
    const float* __restrict__ wsw1, const float* __restrict__ wsb1,
    float* __restrict__ out) {
  extern __shared__ _Float16 lds[];  // [2][WAVES][16][640] halves = 160 KB
  int tid = threadIdx.x;
  int wave = tid >> 5, lane = tid & 31;
  int lrow = lane & 15, lhi = lane >> 4;
  _Float16* myA = lds + (size_t)wave * 16 * KREP;
  _Float16* myB = lds + (size_t)(WAVES_PER_BLOCK + wave) * 16 * KREP;
  int row0 = blockIdx.x * ROWS_PER_BLOCK + wave * 16;

  // Phase 0: span means (cols 0..511) + sinusoidal span feats (cols 512..639)
  for (int row = 0; row < 16; ++row) {
    int g = row0 + row;
    int si = g < n_chunks ? g : n_chunks - 1;   // clamp tail; EXEC stays all-1s
    int st = spans_start[si];
    int ln = spans_len[si];
    float inv = 1.f / (float)ln;
    const float* hi = csum + (size_t)(st + ln) * WDIM2;
    const float* lo = csum + (size_t)st * WDIM2;
    int cbase = lane * 16;
    #pragma unroll
    for (int gq = 0; gq < 2; ++gq) {
      Pack8 o;
      #pragma unroll
      for (int e = 0; e < 8; ++e) {
        int cc = cbase + gq * 8 + e;
        o.h[e] = (_Float16)((hi[cc] - lo[cc]) * inv);
      }
      *(u32x4*)(myA + qoff(row, (cbase >> 3) + gq)) = o.q;
    }
    if (lane < 16) {                      // 128 feat cols = 16 chunks of 8
      int cb = lane * 8;
      int qd = cb >> 5;                   // which of the 4 position scalars
      int j0 = cb & 31;                   // offset within its 32-col PE block
      float x = (qd == 0) ? (float)st
              : (qd == 1) ? (float)(st + ln)
              : (qd == 2) ? (float)ln
              : ((float)st + 0.5f * (float)ln);
      Pack8 o;
      #pragma unroll
      for (int e = 0; e < 8; ++e) {
        int j = j0 + e;
        float f = __expf(-0.5756462732485114f * (float)(j & 15)); // 10000^(-k/16)
        float ang = x * f;
        o.h[e] = (_Float16)((j < 16) ? __sinf(ang) : __cosf(ang));
      }
      *(u32x4*)(myA + qoff(row, 64 + lane)) = o.q;
    }
  }
  __syncthreads();

  gemm_tile(myA, Wt0, 512, b0, myB, lrow, lhi);   // h = relu(mean @ w0 + b0)
  __syncthreads();
  gemm_tile(myB, Wt1, 512, b1, myA, lrow, lhi);   // phrase -> myA[:,0:512]; feats kept
  __syncthreads();

  // GEMM3 (K=640) with fused final projection: score += relu(c+b)*w1[n]
  float acc[8];
  #pragma unroll
  for (int r = 0; r < 8; ++r) acc[r] = 0.f;
  for (int nt = 0; nt < HDIM / 16; ++nt) {
    v8f c = {};
    int n = nt * 16 + lrow;
    const _Float16* wrow = Wt2 + (size_t)n * KREP;
    for (int ks = 0; ks < KREP; ks += 32) {
      FragH a, b;
      int c0 = (ks >> 3) + lhi;
      a.q[0] = *(const u32x4*)(myA + qoff(lrow, c0));
      a.q[1] = *(const u32x4*)(myA + qoff(lrow, c0 + 2));
      b.q[0] = *(const u32x4*)(wrow + ks + lhi * 8);
      b.q[1] = *(const u32x4*)(wrow + ks + lhi * 8 + 16);
      c = __builtin_amdgcn_wmma_f32_16x16x32_f16(false, a.v, false, b.v,
                                                 (short)0, c, false, false);
    }
    float bb = wsb0[n];
    float ww = wsw1[n];
    #pragma unroll
    for (int r = 0; r < 8; ++r) {
      float v = c[r] + bb;
      acc[r] += (v > 0.f ? v : 0.f) * ww;
    }
  }
  // reduce the 16 N-resident lanes of each half-wave
  #pragma unroll
  for (int d = 1; d < 16; d <<= 1) {
    #pragma unroll
    for (int r = 0; r < 8; ++r) acc[r] += __shfl_xor(acc[r], d, 32);
  }
  if (lrow == 0) {
    float bb1 = wsb1[0];
    #pragma unroll
    for (int r = 0; r < 8; ++r) {
      int g = row0 + lhi * 8 + r;
      if (g < n_chunks) out[g] = acc[r] + bb1;
    }
  }
}

// ---------------------------------------------------------------------------
extern "C" void kernel_launch(void* const* d_in, const int* in_sizes, int n_in,
                              void* d_out, int out_size, void* d_ws, size_t ws_size,
                              hipStream_t stream) {
  const int*   sentence    = (const int*)d_in[0];
  const int*   pos_tags    = (const int*)d_in[1];
  const int*   spans_start = (const int*)d_in[2];
  const int*   spans_len   = (const int*)d_in[3];
  const float* We_wrd      = (const float*)d_in[4];
  const float* We_pos      = (const float*)d_in[5];
  const float* dan_w0      = (const float*)d_in[6];
  const float* dan_b0      = (const float*)d_in[7];
  const float* dan_w1      = (const float*)d_in[8];
  const float* dan_b1      = (const float*)d_in[9];
  const float* ws_w0       = (const float*)d_in[10];
  const float* wsb0        = (const float*)d_in[11];
  const float* wsw1        = (const float*)d_in[12];
  const float* wsb1        = (const float*)d_in[13];
  int n_chunks = in_sizes[2];

  char* ws = (char*)d_ws;
  float*    csum = (float*)ws;      ws += (size_t)2049 * 512 * sizeof(float);
  float*    ctot = (float*)ws;      ws += (size_t)16 * 512 * sizeof(float);
  _Float16* Wt0  = (_Float16*)ws;   ws += (size_t)512 * 512 * sizeof(_Float16);
  _Float16* Wt1  = (_Float16*)ws;   ws += (size_t)512 * 512 * sizeof(_Float16);
  _Float16* Wt2  = (_Float16*)ws;   ws += (size_t)512 * KREP * sizeof(_Float16);

  scan_partial<<<16, 512, 0, stream>>>(sentence, pos_tags, We_wrd, We_pos, csum, ctot);
  scan_fix<<<16, 512, 0, stream>>>(csum, ctot);
  int total = 512 * 512 * 2 + 512 * KREP;
  wcvt<<<(total + 255) / 256, 256, 0, stream>>>(dan_w0, dan_w1, ws_w0, Wt0, Wt1, Wt2);

  int nblk = (n_chunks + ROWS_PER_BLOCK - 1) / ROWS_PER_BLOCK;
  size_t ldsBytes = (size_t)2 * WAVES_PER_BLOCK * 16 * KREP * sizeof(_Float16); // 160 KB
  fused_spans<<<nblk, 128, ldsBytes, stream>>>(spans_start, spans_len, n_chunks, csum,
      Wt0, dan_b0, Wt1, dan_b1, Wt2, wsb0, wsw1, wsb1, (float*)d_out);
}
